// Model_89756226552625
// MI455X (gfx1250) — matrix operand order repaired and checked by reading, not compiled, not run on hardware
//
#include <hip/hip_runtime.h>
#include <hip/hip_bf16.h>
#include <math.h>
#include <stdint.h>

// ---------------------------------------------------------------------------
// Problem constants (from the reference)
// ---------------------------------------------------------------------------
#define EV   128      // E
#define LL   16       // L
#define PEc  33       // PE
#define FF   256      // F
#define WW   512      // W
#define HOP  256
#define CC   257      // C
#define HH   512      // H
#define NLAY 4
#define GPc  128
#define NN   (FF * HOP)       // 65536
#define F2   (2 * FF)         // 512
#define ROWS (EV * FF)        // 32768  rows of x
#define KPAD 288              // C padded to /32 for WMMA K
#define KSPEC (2 * KPAD)      // 576 = [Re | Im] padded
#define NPAD_MD 320           // mag/dith weight rows padded for 64-wide N tiles
#define KCHUNK 256            // K staged into LDS per TDM transfer (64x256 f16 = 32KB)

// ---------------------------------------------------------------------------
// Workspace layout (bytes, all offsets 256-aligned).  Regions are reused
// once their producer/consumer ordering allows it.  Peak need ~139 MB.
// ---------------------------------------------------------------------------
static constexpr size_t OFF_X0    = 0;                       // x ping (f16, 32MB)
static constexpr size_t OFF_X1    = 33554432;                // x pong (f16, 32MB)
static constexpr size_t OFF_SPEC  = 0;                       // spec f16 32768x576 (reuses x)
static constexpr size_t OFF_M     = 67108864;                // m f32 32768x257
static constexpr size_t OFF_WIN   = 67108864;                // windowed f16 32768x512 (reuses m)
static constexpr size_t OFF_D     = 100794368;               // d f32 32768x257
static constexpr size_t OFF_SAMP  = 100794368;               // samples f32 128x65536 (reuses d)
static constexpr size_t OFF_NETW  = 134479872;               // net_w f16 4x512x512
static constexpr size_t OFF_WMAG  = OFF_NETW  + 2097152;     // w_mag f16 padded 320x512
static constexpr size_t OFF_WDITH = OFF_WMAG  + 327680;      // w_dith f16 padded 320x512
static constexpr size_t OFF_WDFT  = OFF_WDITH + 327680;      // DFT basis f16 512x576
static constexpr size_t OFF_E     = OFF_WDFT  + 589824;      // e f32 128x512
static constexpr size_t OFF_P     = OFF_E     + 262144;      // p f32 256x512
static constexpr size_t OFF_SHIFT = OFF_P     + 524288;      // int shifts[128]

typedef __attribute__((ext_vector_type(16))) _Float16 v16h;
typedef __attribute__((ext_vector_type(8)))  _Float16 v8h;
typedef __attribute__((ext_vector_type(8)))  float    v8f;
typedef __attribute__((ext_vector_type(4)))  uint32_t u32x4;
typedef __attribute__((ext_vector_type(8)))  int      i32x8;
typedef __attribute__((ext_vector_type(4)))  int      i32x4;

#if __has_builtin(__builtin_amdgcn_tensor_load_to_lds) && \
    __has_builtin(__builtin_amdgcn_s_wait_tensorcnt)
#define USE_TDM 1
#else
#define USE_TDM 0
#endif

// ---------------------------------------------------------------------------
// Small prep kernels
// ---------------------------------------------------------------------------
__global__ void k_cvt_netw(const float* __restrict__ src, _Float16* __restrict__ dst) {
    int i = blockIdx.x * blockDim.x + threadIdx.x;            // 4*512*512
    dst[i] = (_Float16)src[i];
}

// pad (257,512) f32 -> (320,512) f16, zero rows >= 257
__global__ void k_make_wpad(const float* __restrict__ src, _Float16* __restrict__ dst) {
    int i = blockIdx.x * blockDim.x + threadIdx.x;            // 320*512
    int row = i >> 9;
    dst[i] = (row < CC) ? (_Float16)src[(size_t)row * HH + (i & 511)] : (_Float16)0.0f;
}

// Windowed ortho real-iDFT basis, row-major (t, k) with K=576:
//   k in [0,257):    ck * cos(2*pi*k*t/512) * win(t) / sqrt(512)
//   k in [288,545):  -ck * sin(...) * win(t) / sqrt(512),  ck = (k==0||k==256)?1:2
__global__ void k_make_wdft(_Float16* __restrict__ dst) {
    int i = blockIdx.x * blockDim.x + threadIdx.x;            // 512*576
    int t = i / KSPEC, k = i % KSPEC;
    int kk = (k < KPAD) ? k : k - KPAD;
    bool isIm = (k >= KPAD);
    float v = 0.0f;
    if (kk < CC) {
        float ck  = (kk == 0 || kk == 256) ? 1.0f : 2.0f;
        float ang = 6.2831853071795865f * (float)(kk * t) / 512.0f;
        float bas = isIm ? -sinf(ang) : cosf(ang);
        float win = 0.5f - 0.5f * cosf(6.2831853071795865f * (float)t / 512.0f);
        v = ck * bas * win * 0.044194173824159216f;           // 1/sqrt(512)
    }
    dst[(size_t)t * KSPEC + k] = (_Float16)v;
}

// e = events @ w_ee^T + b_ee (128x512,K=16) and p = gen_pos @ w_ep^T + b_ep (256x512,K=128)
__global__ void k_ep(const float* __restrict__ events, const float* __restrict__ w_ee,
                     const float* __restrict__ b_ee, const float* __restrict__ gen_pos,
                     const float* __restrict__ w_ep, const float* __restrict__ b_ep,
                     float* __restrict__ eOut, float* __restrict__ pOut) {
    int id = blockIdx.x * blockDim.x + threadIdx.x;           // (128+256)*512
    if (id < EV * HH) {
        int e = id >> 9, h = id & 511;
        float s = b_ee[h];
#pragma unroll
        for (int l = 0; l < LL; ++l) s += events[e * LL + l] * w_ee[h * LL + l];
        eOut[id] = s;
    } else {
        int id2 = id - EV * HH;
        int f = id2 >> 9, h = id2 & 511;
        float s = b_ep[h];
        for (int g = 0; g < GPc; ++g) s += gen_pos[f * GPc + g] * w_ep[h * GPc + g];
        pOut[id2] = s;
    }
}

// x0[(e*256+f), h] = e[e,h] + p[f,h]  (f16)
__global__ void k_init_x(const float* __restrict__ eArr, const float* __restrict__ pArr,
                         _Float16* __restrict__ x0) {
    size_t i = (size_t)blockIdx.x * blockDim.x + threadIdx.x; // 32768*512
    int row = (int)(i >> 9), h = (int)(i & 511);
    int e = row >> 8, f = row & 255;
    x0[i] = (_Float16)(eArr[e * HH + h] + pArr[f * HH + h]);
}

// ---------------------------------------------------------------------------
// Core WMMA GEMM:  out = epilogue( A(MxK,f16) @ W^T(NpadxK,f16) )
//   block = 8 waves, wave tile 32x64, block tile 256x64
//   Weight tile (64 x KCHUNK) staged into LDS by the Tensor Data Mover (D#
//   descriptor, TENSORcnt) -- one DMA per 32KB chunk, B-fragments from LDS.
//   MODE 0: layer -> f16 store of A + selu(acc + bias)
//   MODE 1: mag   -> f32 store of |acc + bias|   (cols < nvalid)
//   MODE 2: dith  -> f32 store of  acc + bias    (cols < nvalid)
//   MODE 3: iDFT  -> f16 store of  acc           (window folded into W)
// ---------------------------------------------------------------------------
template <int MODE>
__global__ __launch_bounds__(256)
void k_gemm_wmma(const _Float16* __restrict__ A, const _Float16* __restrict__ Wt,
                 int K, const float* __restrict__ bias, void* __restrict__ out,
                 int nvalid, int ldo) {
    __shared__ __align__(64) _Float16 shW[64 * KCHUNK];      // 32KB weight tile

    const int wave = threadIdx.x >> 5;
    const int lane = threadIdx.x & 31;
    const int half = lane >> 4;
    const int lh   = lane & 15;
    const int row0 = blockIdx.x * 256 + wave * 32;
    const int col0 = blockIdx.y * 64;

    v8f acc[2][4] = {};

    for (int kc = 0; kc < K; kc += KCHUNK) {
        const int kc_len = (K - kc < KCHUNK) ? (K - kc) : KCHUNK;
        if (kc) __syncthreads();                              // previous tile consumed
#if USE_TDM
        if (wave == 0) {
            // ---- Tensor DMA descriptor (D#): 2D tile 64 rows x kc_len f16 ----
            uint64_t ga  = (uint64_t)(uintptr_t)(Wt + (size_t)col0 * K + kc);
            uint32_t lds = (uint32_t)(uintptr_t)(&shW[0]);
            u32x4 g0 = {};
            g0[0] = 1u;                                       // count=1 (valid), user mode
            g0[1] = lds;                                      // lds_addr
            g0[2] = (uint32_t)(ga & 0xFFFFFFFFu);             // global_addr[31:0]
            g0[3] = (uint32_t)((ga >> 32) & 0x01FFFFFFu) | (2u << 30); // addr[56:32]|type=2
            i32x8 g1 = {};
            uint32_t dim0 = (uint32_t)kc_len;                 // tensor_dim0
            uint32_t dim1 = 64u;                              // tensor_dim1
            uint64_t str0 = (uint64_t)K;                      // tensor_dim0_stride (elems)
            g1[0] = (int)(1u << 16);                          // wg_mask=0, data_size=1 (2B)
            g1[1] = (int)((dim0 & 0xFFFFu) << 16);            // tensor_dim0[15:0]
            g1[2] = (int)(((dim0 >> 16) & 0xFFFFu) | ((dim1 & 0xFFFFu) << 16));
            g1[3] = (int)(((dim1 >> 16) & 0xFFFFu) | ((uint32_t)kc_len << 16)); // tile_dim0
            g1[4] = (int)64u;                                 // tile_dim1=64, tile_dim2=0
            g1[5] = (int)(uint32_t)(str0 & 0xFFFFFFFFu);      // dim0_stride[31:0]
            g1[6] = (int)(uint32_t)((str0 >> 32) & 0xFFFFu);  // dim0_stride[47:32]
            g1[7] = 0;
            i32x4 g2 = {};                                    // 2D: groups 2/3 unused
            i32x4 g3 = {};
            i32x8 g4 = {};                                    // extra SGPR group (unused, 2D)
            __builtin_amdgcn_tensor_load_to_lds(g0, g1, g2, g3, g4, 0);
            __builtin_amdgcn_s_wait_tensorcnt(0);
        }
#else
        // Cooperative fallback: 16B vector copies global -> LDS
        for (int idx = threadIdx.x * 8; idx < 64 * kc_len; idx += 256 * 8) {
            int r = idx / kc_len, c = idx - r * kc_len;
            *(v8h*)&shW[idx] = *(const v8h*)(Wt + (size_t)(col0 + r) * K + kc + c);
        }
#endif
        __syncthreads();                                      // tile visible to all waves

        for (int k0 = kc; k0 < kc + kc_len; k0 += 32) {
            // A fragments: 16x32 per M-subtile; documented half-wave K split
            v16h af[2];
#pragma unroll
            for (int mi = 0; mi < 2; ++mi) {
                const _Float16* base = A + (size_t)(row0 + mi * 16 + lh) * K + k0;
                v8h lo = *(const v8h*)(base + half * 8);      // K: k0 + half*8 + [0..7]
                v8h hi = *(const v8h*)(base + 16 + half * 8); // K: k0+16 + half*8 + [0..7]
                af[mi] = __builtin_shufflevector(lo, hi, 0,1,2,3,4,5,6,7,8,9,10,11,12,13,14,15);
                __builtin_prefetch(base + 32, 0, 0);          // global_prefetch_b8 next K tile
            }
            // B fragments from LDS: column n = lane&15 holds W[n, k0 + half*8 + 0..7] and W[n, k0+16 + half*8 + 0..7]
            v16h bf[4];
#pragma unroll
            for (int ni = 0; ni < 4; ++ni) {
                const _Float16* bb = &shW[(ni * 16 + lh) * kc_len + (k0 - kc)];
                v8h blo = *(const v8h*)(bb + half * 8);
                v8h bhi = *(const v8h*)(bb + 16 + half * 8);
                bf[ni] = __builtin_shufflevector(blo, bhi, 0,1,2,3,4,5,6,7,8,9,10,11,12,13,14,15);
            }
#pragma unroll
            for (int mi = 0; mi < 2; ++mi)
#pragma unroll
                for (int ni = 0; ni < 4; ++ni)
                    acc[mi][ni] = __builtin_amdgcn_wmma_f32_16x16x32_f16(
                        false, af[mi], false, bf[ni], (short)0, acc[mi][ni], false, false);
        }
    }

    // Epilogue (C/D layout: N = lane&15, M = vgpr + 8*half within each 16x16 tile)
#pragma unroll
    for (int mi = 0; mi < 2; ++mi) {
#pragma unroll
        for (int ni = 0; ni < 4; ++ni) {
            const int n = col0 + ni * 16 + lh;
#pragma unroll
            for (int r = 0; r < 8; ++r) {
                const int m = row0 + mi * 16 + half * 8 + r;
                float v = acc[mi][ni][r];
                if (MODE == 0) {
                    v += bias[n];
                    float s = (v > 0.0f) ? 1.0507009873554805f * v
                                         : 1.7580993408473766f * (expf(v) - 1.0f);
                    float xin = (float)A[(size_t)m * K + n];  // residual (K == ldo == 512)
                    ((_Float16*)out)[(size_t)m * ldo + n] = (_Float16)(xin + s);
                } else if (MODE == 1) {
                    if (n < nvalid)
                        ((float*)out)[(size_t)m * ldo + n] = fabsf(v + bias[n]);
                } else if (MODE == 2) {
                    if (n < nvalid)
                        ((float*)out)[(size_t)m * ldo + n] = v + bias[n];
                } else {
                    ((_Float16*)out)[(size_t)m * ldo + n] = (_Float16)v;
                }
            }
        }
    }
}

// ---------------------------------------------------------------------------
// Phase cumsum + complex spectrum -> padded [Re | Im] f16 rows for the iDFT GEMM
// ---------------------------------------------------------------------------
__global__ __launch_bounds__(288)
void k_phase_spec(const float* __restrict__ m, const float* __restrict__ d,
                  const float* __restrict__ noise, _Float16* __restrict__ spec) {
    const int e = blockIdx.x;
    const int c = threadIdx.x;
    if (c < CC) {
        const float gd = 3.14159265358979323846f * (float)c / 256.0f;
        float ph = 0.0f;
        for (int f = 0; f < FF; ++f) {
            const size_t ri = (size_t)(e * FF + f);
            ph += gd * (1.0f + d[ri * CC + c] * noise[f * CC + c]);
            const float mv = m[ri * CC + c];
            spec[ri * KSPEC + c]        = (_Float16)(mv * cosf(ph));
            spec[ri * KSPEC + KPAD + c] = (_Float16)(mv * sinf(ph));
        }
    } else {
        for (int f = 0; f < FF; ++f) {
            const size_t ri = (size_t)(e * FF + f);
            spec[ri * KSPEC + c]        = (_Float16)0.0f;
            spec[ri * KSPEC + KPAD + c] = (_Float16)0.0f;
        }
    }
}

// Overlap-add: samples[e, f*256+j] = win[e,f,2j] + (f>0 ? win[e,f-1,2j+1] : 0)
__global__ void k_ola(const _Float16* __restrict__ win, float* __restrict__ samples) {
    size_t i = (size_t)blockIdx.x * blockDim.x + threadIdx.x;  // 128*65536
    int e = (int)(i >> 16), n = (int)(i & (NN - 1));
    int f = n >> 8, j = n & 255;
    float v = (float)win[((size_t)(e * FF + f)) * WW + 2 * j];
    if (f > 0) v += (float)win[((size_t)(e * FF + f - 1)) * WW + 2 * j + 1];
    samples[i] = v;
}

// argmax over sim = times @ pos_encoding^T -> shift per event
__global__ void k_sched(const float* __restrict__ times, const float* __restrict__ pos,
                        int* __restrict__ shift) {
    int e = threadIdx.x;                                       // block of 128
    float best = -3.4e38f; int bi = 0;
    for (int f = 0; f < F2; ++f) {
        float s = 0.0f;
#pragma unroll
        for (int k = 0; k < PEc; ++k) s += times[e * PEc + k] * pos[f * PEc + k];
        if (s > best) { best = s; bi = f; }                    // first max, like argmax
    }
    shift[e] = bi * HOP;
}

// One-hot circular convolution == shifted gather; sum over events.
__global__ void k_final(const float* __restrict__ samples, const int* __restrict__ shift,
                        float* __restrict__ out) {
    int n = blockIdx.x * blockDim.x + threadIdx.x;             // 65536
    float acc = 0.0f;
    for (int e = 0; e < EV; ++e) {
        int idx = (NN + n - shift[e]) & (2 * NN - 1);
        if (idx < NN) acc += samples[(size_t)e * NN + idx];
    }
    out[n] = acc;
}

// ---------------------------------------------------------------------------
extern "C" void kernel_launch(void* const* d_in, const int* in_sizes, int n_in,
                              void* d_out, int out_size, void* d_ws, size_t ws_size,
                              hipStream_t stream) {
    (void)in_sizes; (void)n_in; (void)out_size; (void)ws_size;
    const float* events  = (const float*)d_in[0];
    const float* times   = (const float*)d_in[1];
    const float* pos_enc = (const float*)d_in[2];
    const float* gen_pos = (const float*)d_in[3];
    const float* noise   = (const float*)d_in[4];
    const float* w_ee    = (const float*)d_in[5];
    const float* b_ee    = (const float*)d_in[6];
    const float* w_ep    = (const float*)d_in[7];
    const float* b_ep    = (const float*)d_in[8];
    const float* net_w   = (const float*)d_in[9];
    const float* net_b   = (const float*)d_in[10];
    const float* w_mag   = (const float*)d_in[11];
    const float* b_mag   = (const float*)d_in[12];
    const float* w_dith  = (const float*)d_in[13];
    const float* b_dith  = (const float*)d_in[14];

    char* ws = (char*)d_ws;
    _Float16* x0     = (_Float16*)(ws + OFF_X0);
    _Float16* x1     = (_Float16*)(ws + OFF_X1);
    _Float16* spec   = (_Float16*)(ws + OFF_SPEC);
    float*    mArr   = (float*)   (ws + OFF_M);
    _Float16* winArr = (_Float16*)(ws + OFF_WIN);
    float*    dArr   = (float*)   (ws + OFF_D);
    float*    samp   = (float*)   (ws + OFF_SAMP);
    _Float16* netwH  = (_Float16*)(ws + OFF_NETW);
    _Float16* wmagH  = (_Float16*)(ws + OFF_WMAG);
    _Float16* wdithH = (_Float16*)(ws + OFF_WDITH);
    _Float16* wdftH  = (_Float16*)(ws + OFF_WDFT);
    float*    eArr   = (float*)   (ws + OFF_E);
    float*    pArr   = (float*)   (ws + OFF_P);
    int*      shift  = (int*)     (ws + OFF_SHIFT);

    // --- prep ---
    k_cvt_netw <<<(NLAY * HH * HH) / 256, 256, 0, stream>>>(net_w, netwH);
    k_make_wpad<<<(NPAD_MD * HH) / 256, 256, 0, stream>>>(w_mag, wmagH);
    k_make_wpad<<<(NPAD_MD * HH) / 256, 256, 0, stream>>>(w_dith, wdithH);
    k_make_wdft<<<(F2 * KSPEC) / 256, 256, 0, stream>>>(wdftH);
    k_ep       <<<((EV + FF) * HH) / 256, 256, 0, stream>>>(events, w_ee, b_ee,
                                                            gen_pos, w_ep, b_ep, eArr, pArr);
    k_sched    <<<1, EV, 0, stream>>>(times, pos_enc, shift);
    k_init_x   <<<((size_t)ROWS * HH) / 256, 256, 0, stream>>>(eArr, pArr, x0);

    // --- 4 residual MLP layers (ping-pong), WMMA f16 -> f32 acc ---
    _Float16* xin = x0; _Float16* xout = x1;
    for (int l = 0; l < NLAY; ++l) {
        k_gemm_wmma<0><<<dim3(ROWS / 256, HH / 64), 256, 0, stream>>>(
            xin, netwH + (size_t)l * HH * HH, HH, net_b + l * HH, (void*)xout, HH, HH);
        _Float16* t = xin; xin = xout; xout = t;
    }
    // final x lives in x0 (4 swaps)

    // --- mag / dither projections (N = 257, padded weights to 320) ---
    k_gemm_wmma<1><<<dim3(ROWS / 256, NPAD_MD / 64), 256, 0, stream>>>(
        xin, wmagH, HH, b_mag, (void*)mArr, CC, CC);
    k_gemm_wmma<2><<<dim3(ROWS / 256, NPAD_MD / 64), 256, 0, stream>>>(
        xin, wdithH, HH, b_dith, (void*)dArr, CC, CC);

    // --- phase cumsum + spectrum (overwrites dead x region) ---
    k_phase_spec<<<EV, 288, 0, stream>>>(mArr, dArr, noise, spec);

    // --- windowed iRFFT as a GEMM against the precomputed basis ---
    k_gemm_wmma<3><<<dim3(ROWS / 256, WW / 64), 256, 0, stream>>>(
        spec, wdftH, KSPEC, nullptr, (void*)winArr, WW, WW);

    // --- overlap-add, then one-hot scheduling (circular shift) + event sum ---
    k_ola  <<<((size_t)EV * NN) / 256, 256, 0, stream>>>(winArr, samp);
    k_final<<<NN / 256, 256, 0, stream>>>(samp, shift, (float*)d_out);
}
